// GraphConv_42855183680027
// MI455X (gfx1250) — matrix operand order repaired and checked by reading, not compiled, not run on hardware
//
#include <hip/hip_runtime.h>
#include <hip/hip_bf16.h>
#include <math.h>

#define NNODES 50000
#define NEDGES 800000
#define DFEAT 64
#define NEG_SLOPE 0.01f
#define EPSN 1e-12f

typedef __attribute__((ext_vector_type(16))) _Float16 v16h;
typedef __attribute__((ext_vector_type(8)))  float    v8f;

#define WMMA_F16(A, B, C) \
    __builtin_amdgcn_wmma_f32_16x16x32_f16(false, (A), false, (B), (short)0, (C), false, false)

// ---------------------------------------------------------------- zero fill
__global__ void zero_kernel(float* __restrict__ p, int n) {
    int i = blockIdx.x * blockDim.x + threadIdx.x;
    int stride = gridDim.x * blockDim.x;
    for (; i < n; i += stride) p[i] = 0.0f;
}

// ------------------------------------------------------- edge scatter (mean)
// 16 threads per edge: float4 gather + 4x global_atomic_add_f32.
__global__ void __launch_bounds__(256)
aggregate_kernel(const float* __restrict__ x,
                 const long long* __restrict__ src,
                 const long long* __restrict__ dst,
                 float* __restrict__ sums,
                 float* __restrict__ cnt,
                 int addCount) {
    long long idx = (long long)blockIdx.x * blockDim.x + threadIdx.x;
    long long e = idx >> 4;
    if (e >= NEDGES) return;
    int part = (int)(idx & 15);
    int s = (int)src[e];
    int d = (int)dst[e];
    const float4 v = *reinterpret_cast<const float4*>(x + (size_t)s * DFEAT + part * 4);
    float* o = sums + (size_t)d * DFEAT + part * 4;
    atomicAdd(o + 0, v.x);
    atomicAdd(o + 1, v.y);
    atomicAdd(o + 2, v.z);
    atomicAdd(o + 3, v.w);
    if (addCount && part == 0) atomicAdd(cnt + d, 1.0f);
}

// ---------------------------------------------- weight pre-pack (fp32->f16)
// Pack W (row-major [o][k], used as B = W^T) into fragment-ready layout:
// packed[matrix][frag = nt*2+kt][lane][16 halves], 32 contiguous bytes/lane.
// B layout: lane holds col N = nt*16 + L%16; lanes 0-15 K=kt*32+{0..7,16..23},
// lanes 16-31 K=kt*32+{8..15,24..31}.
__global__ void __launch_bounds__(256)
pack_weights_kernel(const float* __restrict__ W1l, const float* __restrict__ W1r,
                    const float* __restrict__ W2l, const float* __restrict__ W2r,
                    _Float16* __restrict__ packed) {
    const float* W = (blockIdx.x == 0) ? W1l : (blockIdx.x == 1) ? W1r
                   : (blockIdx.x == 2) ? W2l : W2r;
    _Float16* out = packed + (size_t)blockIdx.x * 4096;
    const int t = threadIdx.x;
    const int f = t >> 5, lane = t & 31;
    const int nt = f >> 1, kt = f & 1;
    const int o  = nt * 16 + (lane & 15);
    const int k0 = kt * 32 + ((lane >> 4) & 1) * 8;
    const float* p = W + (size_t)o * DFEAT + k0;
    _Float16* q = out + f * 512 + lane * 16;
#pragma unroll
    for (int h = 0; h < 16; ++h) q[h] = (_Float16)p[h < 8 ? h : h + 8];
}

// A fragment (16x32, 16-bit): lane L holds row M=L%16; lanes 0-15 carry
// K = kt*32 + {0..7, 16..23}, lanes 16-31 carry K = kt*32 + {8..15, 24..31}.
__device__ __forceinline__ v16h load_a_frag(const float* __restrict__ base,
                                            int arow, int kt, int lane, float scale) {
    const float* p = base + (size_t)arow * DFEAT + kt * 32 + ((lane >> 4) & 1) * 8;
    v16h a;
#pragma unroll
    for (int h = 0; h < 8; ++h) {
        a[h]     = (_Float16)(p[h]      * scale);
        a[h + 8] = (_Float16)(p[h + 16] * scale);
    }
    return a;
}

__device__ __forceinline__ void store_tile(float* __restrict__ out, int tile, int col,
                                           float bv, v8f acc, int applyLeaky, int lane) {
    const int rbase = tile * 16 + ((lane >> 4) & 1) * 8;
#pragma unroll
    for (int v = 0; v < 8; ++v) {
        float val = acc[v] + bv;
        if (applyLeaky) val = (val >= 0.0f) ? val : NEG_SLOPE * val;
        out[(size_t)(rbase + v) * DFEAT + col] = val;
    }
}

// --------------------------------------------- fused mean + dual GEMM + act
// Packed f16 weights staged once per block into LDS; each wave owns TWO
// 16-row tiles sharing each B fragment (32 WMMAs / wave).
// out[i,:] = (sums[i,:]/max(cnt,1)) @ Wl^T + b + xin[i,:] @ Wr^T  [+leaky]
__global__ void __launch_bounds__(256)
sage_gemm_kernel(const float* __restrict__ sums,
                 const float* __restrict__ cnt,
                 const float* __restrict__ xin,
                 const _Float16* __restrict__ pkWl,
                 const float* __restrict__ bias,
                 const _Float16* __restrict__ pkWr,
                 float* __restrict__ out,
                 int applyLeaky) {
    __shared__ __align__(16) _Float16 ldsW[2][4096];   // 16 KB of WGP LDS
    const int tid = threadIdx.x;

    // Cooperative stage: 2x 8KB packed weights -> LDS (b128 granularity).
    {
        const uint4* g0 = (const uint4*)pkWl;
        const uint4* g1 = (const uint4*)pkWr;
        uint4* s0 = (uint4*)&ldsW[0][0];
        uint4* s1 = (uint4*)&ldsW[1][0];
        s0[tid]       = g0[tid];
        s0[tid + 256] = g0[tid + 256];
        s1[tid]       = g1[tid];
        s1[tid + 256] = g1[tid + 256];
    }
    __syncthreads();

    const int lane = tid & 31;
    const int wave = tid >> 5;
    const int NT    = NNODES / 16;        // 3125 tiles (N % 16 == 0)
    const int NPAIR = (NT + 1) / 2;       // 1563 tile-pairs

    int pw = blockIdx.x * 8 + wave;
    if (pw >= NPAIR) pw = NPAIR - 1;      // clamp: duplicate identical work,
    const int t0 = pw * 2;                // keeps EXEC all-ones for WMMA
    const int t1 = (t0 + 1 < NT) ? t0 + 1 : t0;

    const int r0 = t0 * 16 + (lane & 15);
    const int r1 = t1 * 16 + (lane & 15);
    const float inv0 = 1.0f / fmaxf(cnt[r0], 1.0f);
    const float inv1 = 1.0f / fmaxf(cnt[r1], 1.0f);

    v16h am[2][2], ax[2][2];
    am[0][0] = load_a_frag(sums, r0, 0, lane, inv0);
    am[0][1] = load_a_frag(sums, r0, 1, lane, inv0);
    am[1][0] = load_a_frag(sums, r1, 0, lane, inv1);
    am[1][1] = load_a_frag(sums, r1, 1, lane, inv1);
    ax[0][0] = load_a_frag(xin,  r0, 0, lane, 1.0f);
    ax[0][1] = load_a_frag(xin,  r0, 1, lane, 1.0f);
    ax[1][0] = load_a_frag(xin,  r1, 0, lane, 1.0f);
    ax[1][1] = load_a_frag(xin,  r1, 1, lane, 1.0f);

#pragma unroll
    for (int nt = 0; nt < 4; ++nt) {
        v8f acc0 = {}, acc1 = {};
#pragma unroll
        for (int kt = 0; kt < 2; ++kt) {
            const v16h bl = *(const v16h*)&ldsW[0][(nt * 2 + kt) * 512 + lane * 16];
            acc0 = WMMA_F16(am[0][kt], bl, acc0);
            acc1 = WMMA_F16(am[1][kt], bl, acc1);
            const v16h br = *(const v16h*)&ldsW[1][(nt * 2 + kt) * 512 + lane * 16];
            acc0 = WMMA_F16(ax[0][kt], br, acc0);
            acc1 = WMMA_F16(ax[1][kt], br, acc1);
        }
        const int col = nt * 16 + (lane & 15);
        const float bv = bias[col];
        store_tile(out, t0, col, bv, acc0, applyLeaky, lane);
        store_tile(out, t1, col, bv, acc1, applyLeaky, lane);
    }
}

// ------------------------------------------------------- row L2 normalize
__global__ void __launch_bounds__(256)
normalize_kernel(float* __restrict__ out) {
    int row = blockIdx.x * 8 + (threadIdx.x >> 5);
    if (row >= NNODES) return;
    int lane = threadIdx.x & 31;
    float* p = out + (size_t)row * DFEAT + lane * 2;
    float2 v = *reinterpret_cast<float2*>(p);
    float ss = v.x * v.x + v.y * v.y;
#pragma unroll
    for (int m = 16; m >= 1; m >>= 1) ss += __shfl_xor(ss, m, 32);
    float invn = 1.0f / fmaxf(sqrtf(ss), EPSN);
    v.x *= invn;
    v.y *= invn;
    *reinterpret_cast<float2*>(p) = v;
}

// -------------------------------------------------------------- launcher
extern "C" void kernel_launch(void* const* d_in, const int* in_sizes, int n_in,
                              void* d_out, int out_size, void* d_ws, size_t ws_size,
                              hipStream_t stream) {
    (void)in_sizes; (void)n_in; (void)out_size; (void)ws_size;

    const float*     x   = (const float*)d_in[0];
    const long long* ei  = (const long long*)d_in[1];   // int64 (2, E)
    // d_in[2] = edge_weight: ignored by reference SAGEConv.
    const float*     W1l = (const float*)d_in[3];
    const float*     b1  = (const float*)d_in[4];
    const float*     W1r = (const float*)d_in[5];
    const float*     W2l = (const float*)d_in[6];
    const float*     b2  = (const float*)d_in[7];
    const float*     W2r = (const float*)d_in[8];
    float*           out = (float*)d_out;

    float* ws   = (float*)d_ws;
    float* sums = ws;                               // N*64 f32
    float* cnt  = sums + (size_t)NNODES * DFEAT;    // N    f32 (16B-aligned)
    float* h    = cnt + NNODES;                     // N*64 f32
    _Float16* packed = (_Float16*)(h + (size_t)NNODES * DFEAT); // 4*4096 f16

    const long long* src = ei;           // edge_index[0]
    const long long* dst = ei + NEDGES;  // edge_index[1]

    const long long aggThreads = (long long)NEDGES * 16;
    const int aggBlocks  = (int)((aggThreads + 255) / 256);
    const int NT         = NNODES / 16;                 // 3125
    const int gemmBlocks = (((NT + 1) / 2) + 7) / 8;    // 196
    const int normBlocks = (NNODES + 7) / 8;

    // One-shot weight packing (fp32 -> f16, fragment-ready layout).
    pack_weights_kernel<<<4, 256, 0, stream>>>(W1l, W1r, W2l, W2r, packed);

    // Layer 1
    zero_kernel<<<512, 256, 0, stream>>>(sums, NNODES * DFEAT + NNODES);
    aggregate_kernel<<<aggBlocks, 256, 0, stream>>>(x, src, dst, sums, cnt, 1);
    sage_gemm_kernel<<<gemmBlocks, 256, 0, stream>>>(sums, cnt, x,
                                                     packed + 0, b1, packed + 4096, h, 1);

    // Layer 2 (degree counts reused; only re-zero the sums)
    zero_kernel<<<512, 256, 0, stream>>>(sums, NNODES * DFEAT);
    aggregate_kernel<<<aggBlocks, 256, 0, stream>>>(h, src, dst, sums, cnt, 0);
    sage_gemm_kernel<<<gemmBlocks, 256, 0, stream>>>(sums, cnt, h,
                                                     packed + 8192, b2, packed + 12288, out, 0);

    // F.normalize(p=2, dim=1)
    normalize_kernel<<<normBlocks, 256, 0, stream>>>(out);
}
